// PointNetSegmentation_4801773437702
// MI455X (gfx1250) — matrix operand order repaired and checked by reading, not compiled, not run on hardware
//
#include <hip/hip_runtime.h>
#include <hip/hip_bf16.h>

// ---------------------------------------------------------------------------
// PointNet++ segmentation forward for gfx1250 (MI455X).
// All dense MLP layers run as f16 WMMA (v_wmma_f32_16x16x32_f16) with f32
// accumulation; weights are pre-converted once into fragment-major f16 blobs
// so each wave loads a B fragment with contiguous 128-bit loads per lane.
// ---------------------------------------------------------------------------

typedef __attribute__((ext_vector_type(16))) _Float16 v16h;
typedef __attribute__((ext_vector_type(8)))  float    v8f;

// ---- WMMA fragment helpers (layouts per CDNA5 ISA section 7.12.2) ----------

// 16-bit A matrix 16x32: lane L holds row (L&15); lanes 0-15 cover K 0..7
// (VGPR0-3, pairs) and 16..23 (VGPR4-7); lanes 16-31 cover 8..15 and 24..31.
__device__ __forceinline__ v16h load_Afrag(const _Float16* __restrict__ A,
                                           int ldin, long rowIn0, int kt) {
  int lane = threadIdx.x & 31;
  const _Float16* p = A + (size_t)(rowIn0 + (lane & 15)) * ldin
                        + (size_t)kt * 32 + ((lane >> 4) << 3);
#pragma unroll
  for (int dummy = 0; dummy < 1; ++dummy) {}
  v16h a;
#pragma unroll
  for (int v = 0; v < 8; ++v) {
    int k0 = (v < 4) ? (2 * v) : (16 + 2 * (v - 4));
    a[2 * v]     = p[k0];
    a[2 * v + 1] = p[k0 + 1];
  }
  return a;
}

// B matrix 32x16 fragment-major blob: 512 halves per fragment, lane-major.
// Lane L holds column (L&15); its 16 half slots hold K = (e&7) + 16*(e>>3) + 8*(L>>4).
// The fragmentizer kernel below writes weights in exactly this order.
__device__ __forceinline__ v16h load_Bfrag(const _Float16* __restrict__ Wf,
                                           int NT, int kt, int nt) {
  int lane = threadIdx.x & 31;
  const _Float16* p = Wf + (((size_t)(kt * NT + nt)) << 9) + (lane << 4);
  return *(const v16h*)p;   // 32B per lane -> two global_load_b128
}

// C/D 16x16 f32: VGPR j holds row (j + 8*(lane>>4)), column (lane&15).
template <typename OutT>
__device__ __forceinline__ void store_D(v8f c, OutT* __restrict__ out, int ldout,
                                        long rowOut0, int nt,
                                        const float* __restrict__ bias, int Cout,
                                        bool relu, long rowLimit) {
  int lane = threadIdx.x & 31;
  int n = lane & 15, h = lane >> 4;
  int ch = nt * 16 + n;
  if (ch >= Cout) return;
  float bv = bias ? bias[ch] : 0.f;
#pragma unroll
  for (int j = 0; j < 8; ++j) {
    long r = rowOut0 + j + 8 * h;
    if (r >= rowLimit) continue;
    float v = c[j] + bv;
    if (relu) v = fmaxf(v, 0.f);
    out[(size_t)r * ldout + ch] = (OutT)v;
  }
}

// One wave pushes one 16-row tile through a dense layer: D = relu(A*W + b).
template <typename OutT>
__device__ __forceinline__ void wave_layer(const _Float16* __restrict__ A, int ldin,
                                           long rowIn0,
                                           const _Float16* __restrict__ Wf,
                                           const float* __restrict__ bias,
                                           int KT, int NT, int Cout,
                                           OutT* __restrict__ out, int ldout,
                                           long rowOut0, bool relu, long rowLimit) {
  for (int nt = 0; nt < NT; ++nt) {
    v8f acc = {0.f, 0.f, 0.f, 0.f, 0.f, 0.f, 0.f, 0.f};
    for (int kt = 0; kt < KT; ++kt) {
      v16h a = load_Afrag(A, ldin, rowIn0, kt);
      v16h b = load_Bfrag(Wf, NT, kt, nt);
      acc = __builtin_amdgcn_wmma_f32_16x16x32_f16(
          /*neg_a=*/false, a, /*neg_b=*/false, b,
          /*c_mod=*/(short)0, acc, /*reuse_a=*/false, /*reuse_b=*/false);
    }
    store_D<OutT>(acc, out, ldout, rowOut0, nt, bias, Cout, relu, rowLimit);
  }
}

// ---- weight fragmentizer: f32 [Cin,Cout] (row-major, ld=ldW, rows offset) ->
//      f16 fragment blobs, zero padded to KT*32 x NT*16 -------------------------
__global__ void frag_weights_kernel(const float* __restrict__ W, _Float16* __restrict__ out,
                                    int Cin, int Cout, int ldW, int rowOff,
                                    int KT, int NT) {
  int idx = blockIdx.x * blockDim.x + threadIdx.x;
  int total = KT * NT * 512;
  if (idx >= total) return;
  int frag = idx >> 9;
  int lane = (idx >> 4) & 31;
  int e = idx & 15;
  int kt = frag / NT, nt = frag - kt * NT;
  int k = kt * 32 + (e & 7) + ((e >> 3) << 4) + ((lane >> 4) << 3);
  int n = nt * 16 + (lane & 15);
  float v = 0.f;
  if (k < Cin && n < Cout) v = W[(size_t)(rowOff + k) * ldW + n];
  out[idx] = (_Float16)v;
}

// ---- farthest point sampling (deterministic, start at 0) --------------------
#define FPS_MAXN 8192
__global__ void fps_kernel(const float* __restrict__ pos, int n, int m,
                           int* __restrict__ idxOut) {
  int b = blockIdx.x;
  const float* P = pos + (size_t)b * n * 3;
  int* outp = idxOut + (size_t)b * m;
  __shared__ float dmin[FPS_MAXN];
  __shared__ float rv[256];
  __shared__ int ri[256];
  __shared__ int curIdx;
  int tid = threadIdx.x;
  for (int i = tid; i < n; i += 256) dmin[i] = 3.4e38f;
  if (tid == 0) { outp[0] = 0; curIdx = 0; }
  __syncthreads();
  for (int s = 1; s < m; ++s) {
    int last = curIdx;
    float lx = P[last * 3], ly = P[last * 3 + 1], lz = P[last * 3 + 2];
    float bestv = -1.f; int besti = 0;
    for (int i = tid; i < n; i += 256) {
      float dx = P[i * 3] - lx, dy = P[i * 3 + 1] - ly, dz = P[i * 3 + 2] - lz;
      float d = dx * dx + dy * dy + dz * dz;
      float dm = fminf(dmin[i], d);
      dmin[i] = dm;
      if (dm > bestv) { bestv = dm; besti = i; }
    }
    rv[tid] = bestv; ri[tid] = besti;
    __syncthreads();
    for (int off = 128; off > 0; off >>= 1) {
      if (tid < off) {
        bool take = (rv[tid + off] > rv[tid]) ||
                    (rv[tid + off] == rv[tid] && ri[tid + off] < ri[tid]);
        if (take) { rv[tid] = rv[tid + off]; ri[tid] = ri[tid + off]; }
      }
      __syncthreads();
    }
    if (tid == 0) { curIdx = ri[0]; outp[s] = ri[0]; }
    __syncthreads();
  }
}

__global__ void gather_pos_kernel(const float* __restrict__ src, const int* __restrict__ idx,
                                  float* __restrict__ dst, int M, int nSrc, int total) {
  int id = blockIdx.x * blockDim.x + threadIdx.x;
  if (id >= total) return;
  int b = id / M;
  int si = idx[id];
  const float* s = src + ((size_t)b * nSrc + si) * 3;
  dst[(size_t)id * 3 + 0] = s[0];
  dst[(size_t)id * 3 + 1] = s[1];
  dst[(size_t)id * 3 + 2] = s[2];
}

// ---- ball query: self first, then first (K-1) candidates inside radius -----
__global__ void ballquery_kernel(const float* __restrict__ posQ, int M,
                                 const float* __restrict__ posSrc, int nSrc,
                                 const int* __restrict__ selfIdx, float r2,
                                 int* __restrict__ nbr, int* __restrict__ cnt, int total) {
  int id = blockIdx.x * blockDim.x + threadIdx.x;
  if (id >= total) return;
  int b = id / M;
  const float* qp = posQ + (size_t)id * 3;
  float qx = qp[0], qy = qp[1], qz = qp[2];
  const float* sp = posSrc + (size_t)b * nSrc * 3;
  int self = selfIdx[id];
  int* nb = nbr + (size_t)id * 64;
  nb[0] = self;
  int c = 1;
  for (int i = 0; i < nSrc && c < 64; ++i) {
    if (i == self) continue;
    float ax = sp[i * 3] - qx, ay = sp[i * 3 + 1] - qy, az = sp[i * 3 + 2] - qz;
    float d = ax * ax + ay * ay + az * az;
    if (d <= r2) { nb[c++] = i; }
  }
  cnt[id] = c;
  for (int j = c; j < 64; ++j) nb[j] = self;
}

// ---- SA (PointNetConv) kernel: one query per block, 64 edges, 3-layer MLP,
//      masked max over edges. 4 waves = 4 x 16-edge row tiles. -----------------
struct SAParams {
  const _Float16 *W0, *W1, *W2;
  const float *b0, *b1, *b2;
  int KT0, NT0, C0;
  int KT1, NT1, C1;
  int KT2, NT2, C2;
  int F, KP0;
};

#define SA_KPMAX 160
#define SA_NPMAX 256

__global__ void __launch_bounds__(128) sa_conv_kernel(
    const float* __restrict__ posSrc, const float* __restrict__ posQ,
    int nSrc, int M,
    const float* __restrict__ xSrcF, const _Float16* __restrict__ xSrcH,
    const int* __restrict__ nbr, const int* __restrict__ cnt,
    _Float16* __restrict__ xOut, SAParams P) {
  __shared__ _Float16 bufA[64 * SA_KPMAX];
  __shared__ _Float16 bufB[64 * SA_NPMAX];
  int q = blockIdx.x;
  int b = q / M;
  int tid = threadIdx.x;
  const int* nb = nbr + (size_t)q * 64;
  int qn = cnt[q];
  const float* qp = posQ + (size_t)q * 3;
  float qx = qp[0], qy = qp[1], qz = qp[2];
  const float* SP = posSrc + (size_t)b * nSrc * 3;
  int KP0 = P.KP0, F = P.F;
  if (tid < 64) {
    int e = tid;
    int srcI = nb[e];
    const float* sp = SP + (size_t)srcI * 3;
    _Float16* row = bufA + e * KP0;
    if (xSrcF) {
      const float* xs = xSrcF + ((size_t)b * nSrc + srcI) * F;
      for (int c = 0; c < F; ++c) row[c] = (_Float16)xs[c];
    } else {
      const _Float16* xs = xSrcH + ((size_t)b * nSrc + srcI) * F;
      for (int c = 0; c < F; ++c) row[c] = xs[c];
    }
    row[F]     = (_Float16)(sp[0] - qx);
    row[F + 1] = (_Float16)(sp[1] - qy);
    row[F + 2] = (_Float16)(sp[2] - qz);
    for (int c = F + 3; c < KP0; ++c) row[c] = (_Float16)0.f;
  }
  __syncthreads();
  int w = tid >> 5, r0 = w * 16;
  int ld1 = P.KT1 * 32;   // padded width of layer0 output
  int ld2 = P.KT2 * 32;   // padded width of layer1 output
  wave_layer<_Float16>(bufA, KP0, r0, P.W0, P.b0, P.KT0, P.NT0, P.C0,
                       bufB, ld1, r0, true, 1L << 40);
  __syncthreads();
  wave_layer<_Float16>(bufB, ld1, r0, P.W1, P.b1, P.KT1, P.NT1, P.C1,
                       bufA, ld2, r0, true, 1L << 40);
  __syncthreads();
  wave_layer<_Float16>(bufA, ld2, r0, P.W2, P.b2, P.KT2, P.NT2, P.C2,
                       bufB, P.C2, r0, true, 1L << 40);
  __syncthreads();
  for (int c = tid; c < P.C2; c += 128) {
    float mx = -3.4e38f;
    for (int e = 0; e < qn; ++e) mx = fmaxf(mx, (float)bufB[e * P.C2 + c]);
    xOut[(size_t)q * P.C2 + c] = (_Float16)mx;
  }
}

// ---- SA3 global MLP: rows = B*M2, [x2|pos2] (259 -> 288 pad) -> 256->512->1024
__global__ void __launch_bounds__(64) sa3_kernel(
    const _Float16* __restrict__ x2h, const float* __restrict__ pos2,
    _Float16* __restrict__ h3h,
    const _Float16* W0, const float* b0, const _Float16* W1, const float* b1,
    const _Float16* W2, const float* b2, int totalRows) {
  __shared__ _Float16 bufA[32 * 512];
  __shared__ _Float16 bufB[32 * 512];
  long row0g = (long)blockIdx.x * 32;
  int tid = threadIdx.x;
  for (int i = tid; i < 32 * 288; i += 64) {
    int r = i / 288, c = i - r * 288;
    long rid = row0g + r;
    _Float16 v = (_Float16)0.f;
    if (c < 256) v = x2h[(size_t)rid * 256 + c];
    else if (c < 259) v = (_Float16)pos2[(size_t)rid * 3 + (c - 256)];
    bufA[r * 288 + c] = v;
  }
  __syncthreads();
  int w = tid >> 5, r0 = w * 16;
  wave_layer<_Float16>(bufA, 288, r0, W0, b0, 9, 16, 256, bufB, 256, r0, true, 1L << 40);
  __syncthreads();
  wave_layer<_Float16>(bufB, 256, r0, W1, b1, 8, 32, 512, bufA, 512, r0, true, 1L << 40);
  __syncthreads();
  wave_layer<_Float16>(bufA, 512, r0, W2, b2, 16, 64, 1024,
                       h3h, 1024, row0g + r0, true, (long)totalRows);
}

__global__ void maxpool_kernel(const _Float16* __restrict__ h3h, float* __restrict__ g,
                               int total, int M2) {
  int id = blockIdx.x * blockDim.x + threadIdx.x;
  if (id >= total) return;
  int b = id >> 10, c = id & 1023;
  const _Float16* p = h3h + ((size_t)b * M2) * 1024 + c;
  float mx = -3.4e38f;
  for (int i = 0; i < M2; ++i) mx = fmaxf(mx, (float)p[(size_t)i * 1024]);
  g[id] = mx;
}

// ---- FP3 trick: cvec[b] = g[b] @ Wg + b0 (k=1 interp source is per-cloud
//      global vector, identical for all rows of a cloud) ----------------------
__global__ void __launch_bounds__(32) cvec_kernel(const float* __restrict__ g,
                                                  const _Float16* Wg, const float* b0,
                                                  float* __restrict__ cvec, int Bc) {
  __shared__ _Float16 bufA[16 * 1024];
  int tid = threadIdx.x;
  for (int i = tid; i < 16 * 1024; i += 32) {
    int r = i >> 10, c = i & 1023;
    bufA[i] = (r < Bc) ? (_Float16)g[r * 1024 + c] : (_Float16)0.f;
  }
  __syncthreads();
  wave_layer<float>(bufA, 1024, 0, Wg, b0, 32, 16, 256, cvec, 256, 0, false, (long)Bc);
}

// FP3 point part: y2 = relu(x2 @ Wx + cvec[b]) -> relu(. @ W1 + b1)
__global__ void __launch_bounds__(128) fp3_kernel(
    const _Float16* __restrict__ x2h, const float* __restrict__ cvec,
    const _Float16* Wx, const _Float16* W1, const float* b1,
    _Float16* __restrict__ y2h, int M2) {
  __shared__ _Float16 bufB[64 * 256];
  long row0g = (long)blockIdx.x * 64;
  int b = (int)(row0g / M2);
  int w = threadIdx.x >> 5, r0 = w * 16;
  wave_layer<_Float16>(x2h, 256, row0g + r0, Wx, cvec + (size_t)b * 256,
                       8, 16, 256, bufB, 256, r0, true, 1L << 40);
  __syncthreads();
  wave_layer<_Float16>(bufB, 256, r0, W1, b1, 8, 16, 256,
                       y2h, 256, row0g + r0, true, 1L << 40);
}

// ---- 3-NN search (brute force; stores per-dst indices + normalized weights)
__global__ void knn3_kernel(const float* __restrict__ pdst, int Md,
                            const float* __restrict__ psrc, int Ms,
                            int* __restrict__ kidx, float* __restrict__ kw, int total) {
  int id = blockIdx.x * blockDim.x + threadIdx.x;
  if (id >= total) return;
  int b = id / Md;
  const float* dp = pdst + (size_t)id * 3;
  float dx = dp[0], dy = dp[1], dz = dp[2];
  const float* sp = psrc + (size_t)b * Ms * 3;
  float d0 = 3.4e38f, d1 = 3.4e38f, d2v = 3.4e38f;
  int i0 = 0, i1 = 0, i2 = 0;
  for (int i = 0; i < Ms; ++i) {
    float ax = sp[i * 3] - dx, ay = sp[i * 3 + 1] - dy, az = sp[i * 3 + 2] - dz;
    float d = ax * ax + ay * ay + az * az;
    if (d < d0)      { d2v = d1; i2 = i1; d1 = d0; i1 = i0; d0 = d; i0 = i; }
    else if (d < d1) { d2v = d1; i2 = i1; d1 = d;  i1 = i; }
    else if (d < d2v){ d2v = d;  i2 = i; }
  }
  float w0 = 1.f / fmaxf(d0, 1e-16f);
  float w1 = 1.f / fmaxf(d1, 1e-16f);
  float w2 = 1.f / fmaxf(d2v, 1e-16f);
  float s = w0 + w1 + w2;
  kidx[(size_t)id * 3 + 0] = i0; kidx[(size_t)id * 3 + 1] = i1; kidx[(size_t)id * 3 + 2] = i2;
  kw[(size_t)id * 3 + 0] = w0 / s; kw[(size_t)id * 3 + 1] = w1 / s; kw[(size_t)id * 3 + 2] = w2 / s;
}

// ---- FP2: rows = B*M1; [interp(y2) | x1] (384) -> 256 -> 128 ----------------
__global__ void __launch_bounds__(64) fp2_kernel(
    const _Float16* __restrict__ y2h, const _Float16* __restrict__ x1h,
    const int* __restrict__ kidx, const float* __restrict__ kw,
    const _Float16* W0, const float* b0, const _Float16* W1, const float* b1,
    _Float16* __restrict__ y1h) {
  __shared__ _Float16 bufA[32 * 384];
  __shared__ _Float16 bufB[32 * 256];
  __shared__ int sI[32 * 3];
  __shared__ float sw[32 * 3];
  long row0g = (long)blockIdx.x * 32;
  int b = (int)(row0g / 2048);
  int tid = threadIdx.x;
  for (int i = tid; i < 96; i += 64) {
    sI[i] = kidx[row0g * 3 + i];
    sw[i] = kw[row0g * 3 + i];
  }
  __syncthreads();
  for (int i = tid; i < 32 * 384; i += 64) {
    int r = i / 384, c = i - r * 384;
    long rid = row0g + r;
    float v;
    if (c < 256) {
      long base = (long)b * 512;
      v = sw[r * 3 + 0] * (float)y2h[(size_t)(base + sI[r * 3 + 0]) * 256 + c]
        + sw[r * 3 + 1] * (float)y2h[(size_t)(base + sI[r * 3 + 1]) * 256 + c]
        + sw[r * 3 + 2] * (float)y2h[(size_t)(base + sI[r * 3 + 2]) * 256 + c];
    } else {
      v = (float)x1h[(size_t)rid * 128 + (c - 256)];
    }
    bufA[i] = (_Float16)v;
  }
  __syncthreads();
  int w = tid >> 5, r0 = w * 16;
  wave_layer<_Float16>(bufA, 384, r0, W0, b0, 12, 16, 256, bufB, 256, r0, true, 1L << 40);
  __syncthreads();
  wave_layer<_Float16>(bufB, 256, r0, W1, b1, 8, 8, 128,
                       y1h, 128, row0g + r0, true, 1L << 40);
}

// ---- FP1: rows = B*N; [interp(y1) | x] (131 -> 160 pad) -> 128 ->128 ->128 --
__global__ void __launch_bounds__(128) fp1_kernel(
    const _Float16* __restrict__ y1h, const float* __restrict__ xin,
    const int* __restrict__ kidx, const float* __restrict__ kw,
    const _Float16* W0, const float* b0, const _Float16* W1, const float* b1,
    const _Float16* W2, const float* b2,
    _Float16* __restrict__ y0h) {
  __shared__ _Float16 bufA[64 * 160];
  __shared__ _Float16 bufB[64 * 128];
  __shared__ int sI[64 * 3];
  __shared__ float sw[64 * 3];
  long row0g = (long)blockIdx.x * 64;
  int b = (int)(row0g / 8192);
  int tid = threadIdx.x;
  for (int i = tid; i < 192; i += 128) {
    sI[i] = kidx[row0g * 3 + i];
    sw[i] = kw[row0g * 3 + i];
  }
  __syncthreads();
  for (int i = tid; i < 64 * 160; i += 128) {
    int r = i / 160, c = i - r * 160;
    long rid = row0g + r;
    float v = 0.f;
    if (c < 128) {
      long base = (long)b * 2048;
      v = sw[r * 3 + 0] * (float)y1h[(size_t)(base + sI[r * 3 + 0]) * 128 + c]
        + sw[r * 3 + 1] * (float)y1h[(size_t)(base + sI[r * 3 + 1]) * 128 + c]
        + sw[r * 3 + 2] * (float)y1h[(size_t)(base + sI[r * 3 + 2]) * 128 + c];
    } else if (c < 131) {
      v = xin[(size_t)rid * 3 + (c - 128)];
    }
    bufA[i] = (_Float16)v;
  }
  __syncthreads();
  int w = tid >> 5, r0 = w * 16;
  wave_layer<_Float16>(bufA, 160, r0, W0, b0, 5, 8, 128, bufB, 128, r0, true, 1L << 40);
  __syncthreads();
  wave_layer<_Float16>(bufB, 128, r0, W1, b1, 4, 8, 128, bufA, 128, r0, true, 1L << 40);
  __syncthreads();
  wave_layer<_Float16>(bufA, 128, r0, W2, b2, 4, 8, 128,
                       y0h, 128, row0g + r0, true, 1L << 40);
}

// ---- head: 128 -> relu 128 -> 4 (no relu) -> log_softmax -------------------
__global__ void __launch_bounds__(128) out_kernel(
    const _Float16* __restrict__ y0h,
    const _Float16* W0, const float* b0, const _Float16* W1, const float* b1,
    float* __restrict__ outp) {
  __shared__ _Float16 bufB[64 * 128];
  __shared__ _Float16 bufC[64 * 16];
  long row0g = (long)blockIdx.x * 64;
  int tid = threadIdx.x;
  int w = tid >> 5, r0 = w * 16;
  wave_layer<_Float16>(y0h, 128, row0g + r0, W0, b0, 4, 8, 128, bufB, 128, r0, true, 1L << 40);
  __syncthreads();
  wave_layer<_Float16>(bufB, 128, r0, W1, b1, 4, 1, 4, bufC, 16, r0, false, 1L << 40);
  __syncthreads();
  if (tid < 64) {
    long rid = row0g + tid;
    float v0 = (float)bufC[tid * 16 + 0];
    float v1 = (float)bufC[tid * 16 + 1];
    float v2 = (float)bufC[tid * 16 + 2];
    float v3 = (float)bufC[tid * 16 + 3];
    float m = fmaxf(fmaxf(v0, v1), fmaxf(v2, v3));
    float s = expf(v0 - m) + expf(v1 - m) + expf(v2 - m) + expf(v3 - m);
    float l = m + logf(s);
    outp[(size_t)rid * 4 + 0] = v0 - l;
    outp[(size_t)rid * 4 + 1] = v1 - l;
    outp[(size_t)rid * 4 + 2] = v2 - l;
    outp[(size_t)rid * 4 + 3] = v3 - l;
  }
}

// ---------------------------------------------------------------------------
extern "C" void kernel_launch(void* const* d_in, const int* in_sizes, int n_in,
                              void* d_out, int out_size, void* d_ws, size_t ws_size,
                              hipStream_t stream) {
  (void)in_sizes; (void)n_in; (void)out_size; (void)ws_size;
  const int Bc = 8, N = 8192, K = 64, M1 = 2048, M2 = 512;
  const float* pos = (const float*)d_in[0];
  const float* xin = (const float*)d_in[1];

  char* ws = (char*)d_ws;
  size_t off = 0;
  auto alloc = [&](size_t bytes) -> char* {
    char* p = ws + off;
    off += (bytes + 255) & ~(size_t)255;
    return p;
  };

  // fragmentized weight specs: {input index, Cin, Cout, ldW, rowOffset}
  struct Spec { int inIdx, Cin, Cout, ldW, rowOff; };
  const Spec specs[19] = {
    {2, 6, 64, 64, 0},    {4, 64, 64, 64, 0},    {6, 64, 128, 128, 0},     // SA1
    {8, 131, 128, 128, 0},{10, 128, 128, 128, 0},{12, 128, 256, 256, 0},   // SA2
    {14, 259, 256, 256, 0},{16, 256, 512, 512, 0},{18, 512, 1024, 1024, 0},// SA3
    {20, 1024, 256, 256, 0},{20, 256, 256, 256, 1024},{22, 256, 256, 256, 0}, // FP3 Wg,Wx,L1
    {24, 384, 256, 256, 0},{26, 256, 128, 128, 0},                          // FP2
    {28, 131, 128, 128, 0},{30, 128, 128, 128, 0},{32, 128, 128, 128, 0},   // FP1
    {34, 128, 128, 128, 0},{36, 128, 4, 4, 0},                              // OUT
  };
  _Float16* Wf[19];
  for (int i = 0; i < 19; ++i) {
    int KT = (specs[i].Cin + 31) / 32, NT = (specs[i].Cout + 15) / 16;
    Wf[i] = (_Float16*)alloc((size_t)KT * NT * 512 * sizeof(_Float16));
  }

  int*      idx1 = (int*)alloc((size_t)Bc * M1 * sizeof(int));
  float*    pos1 = (float*)alloc((size_t)Bc * M1 * 3 * sizeof(float));
  int*      idx2 = (int*)alloc((size_t)Bc * M2 * sizeof(int));
  float*    pos2 = (float*)alloc((size_t)Bc * M2 * 3 * sizeof(float));
  int*      nbr1 = (int*)alloc((size_t)Bc * M1 * K * sizeof(int));
  int*      cnt1 = (int*)alloc((size_t)Bc * M1 * sizeof(int));
  int*      nbr2 = (int*)alloc((size_t)Bc * M2 * K * sizeof(int));
  int*      cnt2 = (int*)alloc((size_t)Bc * M2 * sizeof(int));
  _Float16* x1h  = (_Float16*)alloc((size_t)Bc * M1 * 128 * sizeof(_Float16));
  _Float16* x2h  = (_Float16*)alloc((size_t)Bc * M2 * 256 * sizeof(_Float16));
  _Float16* h3h  = (_Float16*)alloc((size_t)Bc * M2 * 1024 * sizeof(_Float16));
  float*    gbuf = (float*)alloc((size_t)Bc * 1024 * sizeof(float));
  float*    cv   = (float*)alloc((size_t)Bc * 256 * sizeof(float));
  _Float16* y2h  = (_Float16*)alloc((size_t)Bc * M2 * 256 * sizeof(_Float16));
  int*      ki2  = (int*)alloc((size_t)Bc * M1 * 3 * sizeof(int));
  float*    kw2  = (float*)alloc((size_t)Bc * M1 * 3 * sizeof(float));
  _Float16* y1h  = (_Float16*)alloc((size_t)Bc * M1 * 128 * sizeof(_Float16));
  int*      ki1  = (int*)alloc((size_t)Bc * N * 3 * sizeof(int));
  float*    kw1  = (float*)alloc((size_t)Bc * N * 3 * sizeof(float));
  _Float16* y0h  = (_Float16*)alloc((size_t)Bc * N * 128 * sizeof(_Float16));

  // 1) fragmentize all weights (tiny; stays L2 resident afterwards)
  for (int i = 0; i < 19; ++i) {
    int KT = (specs[i].Cin + 31) / 32, NT = (specs[i].Cout + 15) / 16;
    int total = KT * NT * 512;
    frag_weights_kernel<<<(total + 255) / 256, 256, 0, stream>>>(
        (const float*)d_in[specs[i].inIdx], Wf[i], specs[i].Cin, specs[i].Cout,
        specs[i].ldW, specs[i].rowOff, KT, NT);
  }

  // 2) FPS level 1 + gather
  fps_kernel<<<Bc, 256, 0, stream>>>(pos, N, M1, idx1);
  gather_pos_kernel<<<(Bc * M1 + 255) / 256, 256, 0, stream>>>(pos, idx1, pos1, M1, N, Bc * M1);

  // 3) SA1
  ballquery_kernel<<<(Bc * M1 + 127) / 128, 128, 0, stream>>>(
      pos1, M1, pos, N, idx1, 0.2f * 0.2f, nbr1, cnt1, Bc * M1);
  SAParams p1 = {Wf[0], Wf[1], Wf[2],
                 (const float*)d_in[3], (const float*)d_in[5], (const float*)d_in[7],
                 1, 4, 64, 2, 4, 64, 2, 8, 128, 3, 32};
  sa_conv_kernel<<<Bc * M1, 128, 0, stream>>>(pos, pos1, N, M1, xin,
                                              (const _Float16*)nullptr, nbr1, cnt1, x1h, p1);

  // 4) FPS level 2 + SA2
  fps_kernel<<<Bc, 256, 0, stream>>>(pos1, M1, M2, idx2);
  gather_pos_kernel<<<(Bc * M2 + 255) / 256, 256, 0, stream>>>(pos1, idx2, pos2, M2, M1, Bc * M2);
  ballquery_kernel<<<(Bc * M2 + 127) / 128, 128, 0, stream>>>(
      pos2, M2, pos1, M1, idx2, 0.4f * 0.4f, nbr2, cnt2, Bc * M2);
  SAParams p2 = {Wf[3], Wf[4], Wf[5],
                 (const float*)d_in[9], (const float*)d_in[11], (const float*)d_in[13],
                 5, 8, 128, 4, 8, 128, 4, 16, 256, 128, 160};
  sa_conv_kernel<<<Bc * M2, 128, 0, stream>>>(pos1, pos2, M1, M2,
                                              (const float*)nullptr, x1h, nbr2, cnt2, x2h, p2);

  // 5) SA3 + global max pool + per-cloud FP3 bias vector
  sa3_kernel<<<(Bc * M2) / 32, 64, 0, stream>>>(
      x2h, pos2, h3h, Wf[6], (const float*)d_in[15], Wf[7], (const float*)d_in[17],
      Wf[8], (const float*)d_in[19], Bc * M2);
  maxpool_kernel<<<(Bc * 1024 + 255) / 256, 256, 0, stream>>>(h3h, gbuf, Bc * 1024, M2);
  cvec_kernel<<<1, 32, 0, stream>>>(gbuf, Wf[9], (const float*)d_in[21], cv, Bc);

  // 6) FP3
  fp3_kernel<<<(Bc * M2) / 64, 128, 0, stream>>>(x2h, cv, Wf[10], Wf[11],
                                                 (const float*)d_in[23], y2h, M2);
  // 7) FP2
  knn3_kernel<<<(Bc * M1 + 127) / 128, 128, 0, stream>>>(pos1, M1, pos2, M2, ki2, kw2, Bc * M1);
  fp2_kernel<<<(Bc * M1) / 32, 64, 0, stream>>>(y2h, x1h, ki2, kw2, Wf[12],
                                                (const float*)d_in[25], Wf[13],
                                                (const float*)d_in[27], y1h);
  // 8) FP1
  knn3_kernel<<<(Bc * N + 127) / 128, 128, 0, stream>>>(pos, N, pos1, M1, ki1, kw1, Bc * N);
  fp1_kernel<<<(Bc * N) / 64, 128, 0, stream>>>(y1h, xin, ki1, kw1,
                                                Wf[14], (const float*)d_in[29],
                                                Wf[15], (const float*)d_in[31],
                                                Wf[16], (const float*)d_in[33], y0h);
  // 9) head + log_softmax
  out_kernel<<<(Bc * N) / 64, 128, 0, stream>>>(y0h, Wf[17], (const float*)d_in[35],
                                                Wf[18], (const float*)d_in[37],
                                                (float*)d_out);
}
